// MultiHeadAttention_79130477461521
// MI455X (gfx1250) — matrix operand order repaired and checked by reading, not compiled, not run on hardware
//
#include <hip/hip_runtime.h>
#include <hip/hip_bf16.h>

#define B_  4
#define S_  2048
#define E_  1024
#define H_  16
#define DH_ 64
#define HD_ (H_*DH_)

typedef __attribute__((ext_vector_type(8)))  _Float16 v8h;
typedef __attribute__((ext_vector_type(16))) _Float16 v16h;
typedef __attribute__((ext_vector_type(8)))  float    v8f;

static __device__ __forceinline__ v8f zero8() {
  v8f z = {0.f,0.f,0.f,0.f,0.f,0.f,0.f,0.f};
  return z;
}

static __device__ __forceinline__ v8f wmma_f16(v16h a, v16h b, v8f c) {
  // D = A(16x32 f16) * B(32x16 f16) + C(16x16 f32)
  return __builtin_amdgcn_wmma_f32_16x16x32_f16(false, a, false, b, (short)0, c, false, false);
}

// A-fragment (16x32 f16) load from a row-major array, row stride in halves.
// rowbase points at element [m][0] for this lane (m = lane&15).
static __device__ __forceinline__ v16h load_a_frag(const _Float16* rowbase, int kb, int halfg) {
  union { v16h v; v8h h[2]; } u;
  u.h[0] = *(const v8h*)(rowbase + kb + halfg*16);
  u.h[1] = *(const v8h*)(rowbase + kb + halfg*16 + 8);
  return u.v;
}

// ---- CDNA5 async global->LDS copy (no VGPR round trip, ASYNCcnt tracked) ----
// GV addressing: 64-bit global address in a VGPR pair; LDS byte offset in a
// VGPR (flat shared pointers carry the wave-relative LDS offset in addr[31:0]).
static __device__ __forceinline__ void async_ld_b128(void* lds, const void* g) {
  unsigned int       la = (unsigned int)(uintptr_t)lds;
  unsigned long long ga = (unsigned long long)(uintptr_t)g;
  asm volatile("global_load_async_to_lds_b128 %0, %1, off"
               :: "v"(la), "v"(ga) : "memory");
}

template<int N>
static __device__ __forceinline__ void wait_asynccnt() {
  asm volatile("s_wait_asynccnt %0" :: "i"(N) : "memory");
}

// ---------------- conversion kernels ----------------

__global__ __launch_bounds__(256) void cvt_f32_to_f16(const float* __restrict__ src,
                                                      _Float16* __restrict__ dst, int n) {
  int i = blockIdx.x*256 + threadIdx.x;
  if (i < n) dst[i] = (_Float16)src[i];
}

// W: [H][E][Dh] -> WT: [H][Dh][E]  (f32 -> f16)
__global__ __launch_bounds__(256) void cvt_w_transpose(const float* __restrict__ W,
                                                       _Float16* __restrict__ WT) {
  int i = blockIdx.x*256 + threadIdx.x;
  if (i >= H_*E_*DH_) return;
  int h   = i / (E_*DH_);
  int rem = i % (E_*DH_);
  int e   = rem / DH_;
  int d   = rem % DH_;
  WT[((size_t)h*DH_ + d)*E_ + e] = (_Float16)W[i];
}

// Wo: [HD][E] -> WoT: [E][HD]  (f32 -> f16)
__global__ __launch_bounds__(256) void cvt_wo_transpose(const float* __restrict__ Wo,
                                                        _Float16* __restrict__ WoT) {
  int i = blockIdx.x*256 + threadIdx.x;
  if (i >= HD_*E_) return;
  int k = i / E_;
  int n = i % E_;
  WoT[(size_t)n*HD_ + k] = (_Float16)Wo[i];
}

// ---------------- QKV projection ----------------
// Per block: one (b,h), 128 rows of S.  8 waves x 16-row strips.  x tile staged
// once per K-step (async, double-buffered) and reused for Q, K and V GEMMs.
// Q,K stored [bh][S][Dh] f16;  V stored transposed [bh][Dh][S] f16.
__global__ __launch_bounds__(256) void qkv_kernel(
    const _Float16* __restrict__ xh,
    const _Float16* __restrict__ WqT, const _Float16* __restrict__ WkT,
    const _Float16* __restrict__ WvT,
    const float* __restrict__ bq, const float* __restrict__ bk,
    const float* __restrict__ bv,
    _Float16* __restrict__ Qh, _Float16* __restrict__ Kh,
    _Float16* __restrict__ Vth) {
  const int st = blockIdx.x;
  const int h  = blockIdx.y;
  const int b  = blockIdx.z;
  const int bh = b*H_ + h;
  const int tid   = threadIdx.x;
  const int wave  = tid >> 5;
  const int lane  = tid & 31;
  const int halfg = lane >> 4;
  const int n     = lane & 15;
  const int s0    = st*128;

  __shared__ __align__(32) _Float16 xs[2][128*32];
  __shared__ __align__(32) _Float16 wqs[2][64*32];
  __shared__ __align__(32) _Float16 wks[2][64*32];
  __shared__ __align__(32) _Float16 wvs[2][64*32];

  v8f aq[4], ak[4], av[4];
  for (int j = 0; j < 4; ++j) { aq[j] = zero8(); ak[j] = zero8(); av[j] = zero8(); }

  const size_t wbase = (size_t)h*DH_*E_;
  const int xrow = tid >> 1, xcol = (tid & 1)*16;   // 32B of x tile per thread
  const int wrow = tid >> 2, wcol = (tid & 3)*8;    // 16B of each W tile per thread

  // 5 async b128 loads per thread per K-step
  auto issue = [&](int e0, int buf) {
    _Float16* lx = &xs[buf][xrow*32 + xcol];
    const _Float16* gx = &xh[((size_t)(b*S_ + s0 + xrow))*E_ + e0 + xcol];
    async_ld_b128(lx, gx);
    async_ld_b128(lx + 8, gx + 8);
    async_ld_b128(&wqs[buf][wrow*32 + wcol], &WqT[wbase + (size_t)wrow*E_ + e0 + wcol]);
    async_ld_b128(&wks[buf][wrow*32 + wcol], &WkT[wbase + (size_t)wrow*E_ + e0 + wcol]);
    async_ld_b128(&wvs[buf][wrow*32 + wcol], &WvT[wbase + (size_t)wrow*E_ + e0 + wcol]);
  };

  issue(0, 0);
  const int nsteps = E_/32;
  for (int step = 0; step < nsteps; ++step) {
    const int buf = step & 1;
    if (step + 1 < nsteps) { issue((step + 1)*32, buf ^ 1); wait_asynccnt<5>(); }
    else                   { wait_asynccnt<0>(); }
    __syncthreads();

    v16h a = load_a_frag(&xs[buf][(wave*16 + n)*32], 0, halfg);
    for (int j = 0; j < 4; ++j) {
      v16h bq_ = *(const v16h*)&wqs[buf][(j*16 + n)*32 + halfg*16];
      aq[j] = wmma_f16(a, bq_, aq[j]);
      v16h bk_ = *(const v16h*)&wks[buf][(j*16 + n)*32 + halfg*16];
      ak[j] = wmma_f16(a, bk_, ak[j]);
      v16h bv_ = *(const v16h*)&wvs[buf][(j*16 + n)*32 + halfg*16];
      av[j] = wmma_f16(a, bv_, av[j]);
    }
    __syncthreads();
  }

  for (int j = 0; j < 4; ++j) {
    const int d = j*16 + n;
    const float bqv = bq[h*DH_ + d];
    const float bkv = bk[h*DH_ + d];
    const float bvv = bv[h*DH_ + d];
    for (int r = 0; r < 8; ++r) {
      const int s = s0 + wave*16 + r + 8*halfg;     // C-layout: M = r + 8*half
      Qh[((size_t)bh*S_ + s)*DH_ + d]  = (_Float16)(aq[j][r] + bqv);
      Kh[((size_t)bh*S_ + s)*DH_ + d]  = (_Float16)(ak[j][r] + bkv);
      Vth[((size_t)bh*DH_ + d)*S_ + s] = (_Float16)(av[j][r] + bvv);
    }
  }
}

// ---------------- flash attention (causal) ----------------
// Per block: one (b,h), 128 queries.  Each of 8 waves owns a 16-query strip so
// online-softmax row stats only need 16-lane shuffles.  K / V^T tiles of 64
// keys are staged async + double-buffered in LDS and shared by all waves.
__global__ __launch_bounds__(256) void attn_kernel(
    const _Float16* __restrict__ Qh, const _Float16* __restrict__ Kh,
    const _Float16* __restrict__ Vth, _Float16* __restrict__ ctxh) {
  const int qt = blockIdx.x;
  const int h  = blockIdx.y;
  const int b  = blockIdx.z;
  const int bh = b*H_ + h;
  const int tid   = threadIdx.x;
  const int wave  = tid >> 5;
  const int lane  = tid & 31;
  const int halfg = lane >> 4;
  const int n     = lane & 15;
  const int q0    = qt*128 + wave*16;

  __shared__ __align__(32) _Float16 ks[2][64*64];    // [k][d]  (B for Q*K^T)
  __shared__ __align__(32) _Float16 vts[2][64*64];   // [d][k]  (B for P*V)
  __shared__ __align__(32) _Float16 ps[8][16*64];    // per-wave P strip

  const _Float16* qrow = Qh + ((size_t)bh*S_ + q0 + n)*DH_;
  v16h qa0 = load_a_frag(qrow, 0, halfg);
  v16h qa1 = load_a_frag(qrow, 32, halfg);

  v8f ctx[4];
  for (int j = 0; j < 4; ++j) ctx[j] = zero8();
  float mrow[8], lrow[8];
  for (int r = 0; r < 8; ++r) { mrow[r] = -1e30f; lrow[r] = 0.f; }

  const int row = tid >> 2, col = (tid & 3)*16;      // 32B of each tile per thread

  // 4 async b128 loads per thread per key tile
  auto issue = [&](int kt, int buf) {
    const int k0 = kt*64;
    _Float16* lk = &ks[buf][row*64 + col];
    const _Float16* gk = &Kh[((size_t)bh*S_ + k0 + row)*DH_ + col];
    async_ld_b128(lk, gk);
    async_ld_b128(lk + 8, gk + 8);
    _Float16* lv = &vts[buf][row*64 + col];
    const _Float16* gv = &Vth[((size_t)bh*DH_ + row)*S_ + k0 + col];
    async_ld_b128(lv, gv);
    async_ld_b128(lv + 8, gv + 8);
  };

  const int nkt = (qt + 1)*2;                        // causal: keys < 128*(qt+1)
  issue(0, 0);
  for (int kt = 0; kt < nkt; ++kt) {
    const int k0  = kt*64;
    const int buf = kt & 1;
    if (kt + 1 < nkt) { issue(kt + 1, buf ^ 1); wait_asynccnt<4>(); }
    else              { wait_asynccnt<0>(); }
    __syncthreads();

    v8f sc[4];
    for (int j = 0; j < 4; ++j) sc[j] = zero8();
    for (int kc = 0; kc < 2; ++kc) {
      v16h a = kc ? qa1 : qa0;
      for (int j = 0; j < 4; ++j) {
        v16h bf = *(const v16h*)&ks[buf][(j*16 + n)*64 + kc*32 + halfg*16];
        sc[j] = wmma_f16(a, bf, sc[j]);
      }
    }

    // online softmax per row (row M = r + 8*half lives in one 16-lane group)
    for (int r = 0; r < 8; ++r) {
      const int q = q0 + r + 8*halfg;
      float mx = -1e30f;
      for (int j = 0; j < 4; ++j) {
        const int k = k0 + j*16 + n;
        float v = (k <= q) ? sc[j][r]*0.125f : -1e30f;   // 1/sqrt(64)
        sc[j][r] = v;
        mx = fmaxf(mx, v);
      }
      for (int o = 1; o < 16; o <<= 1) mx = fmaxf(mx, __shfl_xor(mx, o, 32));
      const float mnew = fmaxf(mrow[r], mx);
      const float corr = __expf(mrow[r] - mnew);
      mrow[r] = mnew;
      lrow[r] *= corr;
      for (int j = 0; j < 4; ++j) ctx[j][r] *= corr;
      float rs = 0.f;
      for (int j = 0; j < 4; ++j) {
        const float v = sc[j][r];
        const float p = (v > -1e29f) ? __expf(v - mnew) : 0.f;
        rs += p;
        ps[wave][(r + 8*halfg)*64 + j*16 + n] = (_Float16)p;
      }
      for (int o = 1; o < 16; o <<= 1) rs += __shfl_xor(rs, o, 32);
      lrow[r] += rs;
    }

    // ctx += P(16x64) @ V(64x64)   (P re-read in A layout; wave-local LDS)
    for (int kc = 0; kc < 2; ++kc) {
      v16h pa = load_a_frag(&ps[wave][n*64], kc*32, halfg);
      for (int j = 0; j < 4; ++j) {
        v16h vb = *(const v16h*)&vts[buf][(j*16 + n)*64 + kc*32 + halfg*16];
        ctx[j] = wmma_f16(pa, vb, ctx[j]);
      }
    }
    __syncthreads();
  }

  for (int r = 0; r < 8; ++r) {
    const float inv = 1.0f / lrow[r];
    const int s = q0 + r + 8*halfg;
    for (int j = 0; j < 4; ++j) {
      const int d = j*16 + n;
      ctxh[((size_t)(b*S_ + s))*HD_ + h*DH_ + d] = (_Float16)(ctx[j][r]*inv);
    }
  }
}

// ---------------- output projection ----------------
// out[B*S, E] = ctx_h[B*S, HD] @ Wo[HD, E] + bo  (fp32 epilogue)
__global__ __launch_bounds__(256) void outproj_kernel(
    const _Float16* __restrict__ ctxh, const _Float16* __restrict__ WoT,
    const float* __restrict__ bo, float* __restrict__ out) {
  const int mt = blockIdx.x;      // 128-row tiles over B*S
  const int nt = blockIdx.y;      // 64-col tiles over E
  const int tid   = threadIdx.x;
  const int wave  = tid >> 5;
  const int lane  = tid & 31;
  const int halfg = lane >> 4;
  const int n     = lane & 15;

  __shared__ __align__(32) _Float16 as[2][128*32];
  __shared__ __align__(32) _Float16 bs[2][64*32];

  v8f acc[4];
  for (int j = 0; j < 4; ++j) acc[j] = zero8();

  const int arow = tid >> 1, acol = (tid & 1)*16;
  const int wrow = tid >> 2, wcol = (tid & 3)*8;

  // 3 async b128 loads per thread per K-step
  auto issue = [&](int e0, int buf) {
    _Float16* la = &as[buf][arow*32 + acol];
    const _Float16* ga = &ctxh[((size_t)(mt*128 + arow))*HD_ + e0 + acol];
    async_ld_b128(la, ga);
    async_ld_b128(la + 8, ga + 8);
    async_ld_b128(&bs[buf][wrow*32 + wcol],
                  &WoT[((size_t)(nt*64 + wrow))*HD_ + e0 + wcol]);
  };

  issue(0, 0);
  const int nsteps = HD_/32;
  for (int step = 0; step < nsteps; ++step) {
    const int buf = step & 1;
    if (step + 1 < nsteps) { issue((step + 1)*32, buf ^ 1); wait_asynccnt<3>(); }
    else                   { wait_asynccnt<0>(); }
    __syncthreads();

    v16h a = load_a_frag(&as[buf][(wave*16 + n)*32], 0, halfg);
    for (int j = 0; j < 4; ++j) {
      v16h bf = *(const v16h*)&bs[buf][(j*16 + n)*32 + halfg*16];
      acc[j] = wmma_f16(a, bf, acc[j]);
    }
    __syncthreads();
  }

  for (int j = 0; j < 4; ++j) {
    const int ncol = nt*64 + j*16 + n;
    const float bias = bo[ncol];
    for (int r = 0; r < 8; ++r) {
      const int row = mt*128 + wave*16 + r + 8*halfg;
      out[(size_t)row*E_ + ncol] = acc[j][r] + bias;
    }
  }
}

// ---------------- launcher ----------------
extern "C" void kernel_launch(void* const* d_in, const int* in_sizes, int n_in,
                              void* d_out, int out_size, void* d_ws, size_t ws_size,
                              hipStream_t stream) {
  const float* x  = (const float*)d_in[0];
  const float* Wq = (const float*)d_in[1];
  const float* bq = (const float*)d_in[2];
  const float* Wk = (const float*)d_in[3];
  const float* bk = (const float*)d_in[4];
  const float* Wv = (const float*)d_in[5];
  const float* bv = (const float*)d_in[6];
  const float* Wo = (const float*)d_in[7];
  const float* bo = (const float*)d_in[8];
  float* out = (float*)d_out;

  char* ws = (char*)d_ws;
  size_t off = 0;
  auto alloc = [&](size_t bytes) -> void* {
    void* p = ws + off;
    off += (bytes + 255) & ~(size_t)255;
    return p;
  };
  _Float16* xh   = (_Float16*)alloc((size_t)B_*S_*E_*2);       // 16 MB
  _Float16* Qh   = (_Float16*)alloc((size_t)B_*H_*S_*DH_*2);   // 16 MB
  _Float16* Kh   = (_Float16*)alloc((size_t)B_*H_*S_*DH_*2);   // 16 MB
  _Float16* Vth  = (_Float16*)alloc((size_t)B_*H_*DH_*S_*2);   // 16 MB
  _Float16* ctxh = (_Float16*)alloc((size_t)B_*S_*HD_*2);      // 16 MB
  _Float16* WqT  = (_Float16*)alloc((size_t)H_*DH_*E_*2);      //  2 MB
  _Float16* WkT  = (_Float16*)alloc((size_t)H_*DH_*E_*2);
  _Float16* WvT  = (_Float16*)alloc((size_t)H_*DH_*E_*2);
  _Float16* WoT  = (_Float16*)alloc((size_t)E_*HD_*2);

  const int nx = B_*S_*E_;
  cvt_f32_to_f16<<<(nx + 255)/256, 256, 0, stream>>>(x, xh, nx);
  cvt_w_transpose<<<(H_*E_*DH_ + 255)/256, 256, 0, stream>>>(Wq, WqT);
  cvt_w_transpose<<<(H_*E_*DH_ + 255)/256, 256, 0, stream>>>(Wk, WkT);
  cvt_w_transpose<<<(H_*E_*DH_ + 255)/256, 256, 0, stream>>>(Wv, WvT);
  cvt_wo_transpose<<<(HD_*E_ + 255)/256, 256, 0, stream>>>(Wo, WoT);

  qkv_kernel<<<dim3(S_/128, H_, B_), 256, 0, stream>>>(
      xh, WqT, WkT, WvT, bq, bk, bv, Qh, Kh, Vth);

  attn_kernel<<<dim3(S_/128, H_, B_), 256, 0, stream>>>(Qh, Kh, Vth, ctxh);

  outproj_kernel<<<dim3((B_*S_)/128, E_/64), 256, 0, stream>>>(ctxh, WoT, bo, out);
}
